// PerformerAttention_56169582297504
// MI455X (gfx1250) — matrix operand order repaired and checked by reading, not compiled, not run on hardware
//
#include <hip/hip_runtime.h>
#include <hip/hip_bf16.h>
#include <math.h>

#define B_  8
#define N_  4096
#define C_  768
#define H_  12
#define DH_ 64
#define M_  256
#define BH_ (B_ * H_)
#define BN_ (B_ * N_)

typedef __attribute__((ext_vector_type(16))) __bf16 bfrag16;
typedef __attribute__((ext_vector_type(8)))  float  v8f;
typedef __attribute__((ext_vector_type(4)))  int    v4i_;

#define GAS __attribute__((address_space(1)))
#define LAS __attribute__((address_space(3)))

#if defined(__has_builtin)
#if __has_builtin(__builtin_amdgcn_global_load_async_to_lds_b128) && \
    __has_builtin(__builtin_amdgcn_s_wait_asynccnt)
#define HAVE_ASYNC_LDS 1
#endif
#endif

__device__ __forceinline__ int lane_() { return threadIdx.x & 31; }

__device__ __forceinline__ v8f zero8() {
  v8f z;
#pragma unroll
  for (int i = 0; i < 8; ++i) z[i] = 0.0f;
  return z;
}

__device__ __forceinline__ v8f wmma_bf16(bfrag16 a, bfrag16 b, v8f c) {
  // D = A(16x32 bf16) * B(32x16 bf16) + C(16x16 f32)
  return __builtin_amdgcn_wmma_f32_16x16x32_bf16(false, a, false, b, (short)0, c,
                                                 false, false);
}

// A fragment (16x32 bf16) from row-major bf16 at (row0,k0), leading dim ld.
// Lane layout: row = lane%16; elem e -> K=(lane/16)*16+e.
__device__ __forceinline__ bfrag16 load_a_bf16(const __bf16* p, int ld) {
  int l = lane_();
  const __bf16* row = p + (size_t)(l & 15) * ld + ((l >> 4) << 4);
  bfrag16 a;
#pragma unroll
  for (int i = 0; i < 8; ++i) a[i] = row[i];
#pragma unroll
  for (int i = 0; i < 8; ++i) a[8 + i] = row[8 + i];
  return a;
}

// Same but converting from fp32 source.
__device__ __forceinline__ bfrag16 load_a_f32(const float* p, int ld) {
  int l = lane_();
  const float* row = p + (size_t)(l & 15) * ld + ((l >> 4) << 4);
  bfrag16 a;
#pragma unroll
  for (int i = 0; i < 8; ++i) a[i] = (__bf16)row[i];
#pragma unroll
  for (int i = 0; i < 8; ++i) a[8 + i] = (__bf16)row[8 + i];
  return a;
}

// B fragment (32x16) from transposed storage BT[ncol][k] (contiguous in k).
// Lane layout: col = lane%16, K = (lane/16)*16 + e.
__device__ __forceinline__ bfrag16 load_bT(const __bf16* p, int ld) {
  int l = lane_();
  const __bf16* cp = p + (size_t)(l & 15) * ld + ((l >> 4) << 4);
  bfrag16 b;
#pragma unroll
  for (int i = 0; i < 16; ++i) b[i] = cp[i];
  return b;
}

// B fragment (32x16) from row-major storage Brm[k][ncol].
__device__ __forceinline__ bfrag16 load_b_rm(const __bf16* p, int ld) {
  int l = lane_();
  int col = l & 15;
  int kb = (l >> 4) << 4;
  bfrag16 b;
#pragma unroll
  for (int i = 0; i < 16; ++i) b[i] = p[(size_t)(kb + i) * ld + col];
  return b;
}

// ---------------- kernel 0: weight conversion ----------------
// in: [K][J] f32 row-major; out: [J][K] bf16 (transposed)
__global__ void perf_transpose_bf16(const float* __restrict__ in,
                                    __bf16* __restrict__ out, int K, int J) {
  int idx = blockIdx.x * 256 + threadIdx.x;
  if (idx >= K * J) return;
  int j = idx / K, k = idx - j * K;
  out[idx] = (__bf16)in[(size_t)k * J + j];
}

__global__ void perf_convert_bf16(const float* __restrict__ in,
                                  __bf16* __restrict__ out, int n) {
  int idx = blockIdx.x * 256 + threadIdx.x;
  if (idx < n) out[idx] = (__bf16)in[idx];
}

// ---------------- kernel 1: QKV GEMM (ping-pong double buffered) ----------------
// x [BN, C] f32 ; WtQKV [3C][C] bf16 (transposed) ; qkv 3 planes [BH][N][DH] bf16
// Block = 8 waves as 4(row) x 2(col): block tile 64 rows x 128 cols. C_ % 64 == 0.
__global__ __launch_bounds__(256) void perf_qkv_gemm(
    const float* __restrict__ x, const __bf16* __restrict__ WtQKV,
    __bf16* __restrict__ qkv) {
  int w = threadIdx.x >> 5, l = threadIdx.x & 31;
  int rbase = blockIdx.y * 64 + (w >> 1) * 16;
  int cbase = blockIdx.x * 128 + (w & 1) * 64;
  const float* xrow = x + (size_t)rbase * C_;
  const __bf16* Wc[4];
#pragma unroll
  for (int t = 0; t < 4; ++t) Wc[t] = WtQKV + (size_t)(cbase + t * 16) * C_;

  v8f acc[4];
#pragma unroll
  for (int t = 0; t < 4; ++t) acc[t] = zero8();

  // ping-pong buffers: set0 holds kk, set1 holds kk+32
  bfrag16 a0 = load_a_f32(xrow, C_);
  bfrag16 b0[4];
#pragma unroll
  for (int t = 0; t < 4; ++t) b0[t] = load_bT(Wc[t], C_);

  for (int kk = 0; kk < C_; kk += 64) {
    __builtin_prefetch(xrow + kk + 64, 0, 3);  // global_prefetch
    bfrag16 a1 = load_a_f32(xrow + kk + 32, C_);
    bfrag16 b1[4];
#pragma unroll
    for (int t = 0; t < 4; ++t) b1[t] = load_bT(Wc[t] + kk + 32, C_);
#pragma unroll
    for (int t = 0; t < 4; ++t) acc[t] = wmma_bf16(a0, b0[t], acc[t]);
    if (kk + 64 < C_) {
      a0 = load_a_f32(xrow + kk + 64, C_);
#pragma unroll
      for (int t = 0; t < 4; ++t) b0[t] = load_bT(Wc[t] + kk + 64, C_);
    }
#pragma unroll
    for (int t = 0; t < 4; ++t) acc[t] = wmma_bf16(a1, b1[t], acc[t]);
  }
  int rsub = (l >> 4) * 8, csub = l & 15;
#pragma unroll
  for (int t = 0; t < 4; ++t) {
    int col = cbase + t * 16 + csub;
    int which = col / C_;
    int rem = col - which * C_;
    int h = rem >> 6, d = rem & 63;
    __bf16* plane = qkv + (size_t)which * BH_ * N_ * DH_;
#pragma unroll
    for (int i = 0; i < 8; ++i) {
      int row = rbase + rsub + i;
      int b2 = row >> 12, n = row & (N_ - 1);
      plane[(((size_t)(b2 * H_ + h)) * N_ + n) * DH_ + d] = (__bf16)acc[t][i];
    }
  }
}

// ---------------- kernel 2: phi(k) -> kv, k_sum (per b,h) ----------------
__global__ __launch_bounds__(256) void perf_kv_ksum(
    const __bf16* __restrict__ kmat, const __bf16* __restrict__ vmat,
    const __bf16* __restrict__ rfb, __bf16* __restrict__ kvws,
    float* __restrict__ ksumws) {
  int bh = blockIdx.x;
  int tid = threadIdx.x, w = tid >> 5, l = tid & 31;
  __shared__ __bf16 s_kpT[M_ * 32];  // [m][n_loc] transposed k_p chunk
  __shared__ __bf16 s_v[32 * DH_];   // [n_loc][d]
  __shared__ float  s_nsq[32];
  const __bf16* kb = kmat + (size_t)bh * N_ * DH_;
  const __bf16* vb = vmat + (size_t)bh * N_ * DH_;
  const float invSqrtM = 0.0625f;  // 1/sqrt(256)

  v8f kvacc[8];
#pragma unroll
  for (int t = 0; t < 8; ++t) kvacc[t] = zero8();
  float ksum = 0.0f;

  for (int n0 = 0; n0 < N_; n0 += 32) {
    // stage v chunk (32x64 bf16 = 4 KB): one 16-byte async transfer per lane
#if defined(HAVE_ASYNC_LDS)
    {
      __bf16* gsrc = const_cast<__bf16*>(vb) + (size_t)n0 * DH_ + tid * 8;
      __builtin_amdgcn_global_load_async_to_lds_b128(
          (GAS v4i_*)gsrc, (LAS v4i_*)(s_v + tid * 8), 0, 0);
    }
#else
    for (int i = tid; i < 32 * DH_; i += 256) s_v[i] = vb[(size_t)n0 * DH_ + i];
#endif
    if (tid < 32) {
      float s = 0.0f;
      const __bf16* kr = kb + (size_t)(n0 + tid) * DH_;
      for (int d = 0; d < DH_; ++d) { float t = (float)kr[d]; s += t * t; }
      s_nsq[tid] = 0.5f * s;
    }
#if defined(HAVE_ASYNC_LDS)
    __builtin_amdgcn_s_wait_asynccnt(0);
#endif
    __syncthreads();
    // phase A: proj = k @ rf^T (K=64), exp, write transposed into LDS
#pragma unroll
    for (int ns = 0; ns < 2; ++ns) {
      bfrag16 a0 = load_a_bf16(kb + (size_t)(n0 + ns * 16) * DH_ + 0, DH_);
      bfrag16 a1 = load_a_bf16(kb + (size_t)(n0 + ns * 16) * DH_ + 32, DH_);
#pragma unroll
      for (int mi = 0; mi < 2; ++mi) {
        int mt = 2 * w + mi;
        bfrag16 b0 = load_bT(rfb + (size_t)(mt * 16) * DH_ + 0, DH_);
        bfrag16 b1 = load_bT(rfb + (size_t)(mt * 16) * DH_ + 32, DH_);
        v8f c = wmma_bf16(a0, b0, zero8());
        c = wmma_bf16(a1, b1, c);
        int rl = ns * 16 + (l >> 4) * 8;
        int mc = mt * 16 + (l & 15);
#pragma unroll
        for (int i = 0; i < 8; ++i) {
          float val = __expf(c[i] - s_nsq[rl + i]) * invSqrtM;
          s_kpT[mc * 32 + rl + i] = (__bf16)val;
        }
      }
    }
    __syncthreads();
    // k_sum: thread tid owns m = tid
    {
      const __bf16* row = s_kpT + tid * 32;
      float s = 0.0f;
#pragma unroll
      for (int j = 0; j < 32; ++j) s += (float)row[j];
      ksum += s;
    }
    // phase B: kv += k_p^T(16x32) @ v(32x16)
#pragma unroll
    for (int mi = 0; mi < 2; ++mi) {
      int mt = 2 * w + mi;
      bfrag16 a = load_a_bf16(s_kpT + mt * 16 * 32, 32);
      bfrag16 b[4];
#pragma unroll
      for (int dt = 0; dt < 4; ++dt) b[dt] = load_b_rm(s_v + dt * 16, DH_);
#pragma unroll
      for (int dt = 0; dt < 4; ++dt)
        kvacc[mi * 4 + dt] = wmma_bf16(a, b[dt], kvacc[mi * 4 + dt]);
    }
    __syncthreads();
  }
  __bf16* kvb = kvws + (size_t)bh * M_ * DH_;
  int rsub = (l >> 4) * 8, csub = l & 15;
#pragma unroll
  for (int mi = 0; mi < 2; ++mi)
#pragma unroll
    for (int dt = 0; dt < 4; ++dt) {
      int m0 = (2 * w + mi) * 16 + rsub;
      int dcol = dt * 16 + csub;
#pragma unroll
      for (int i = 0; i < 8; ++i)
        kvb[(size_t)(m0 + i) * DH_ + dcol] = (__bf16)kvacc[mi * 4 + dt][i];
    }
  ksumws[(size_t)bh * M_ + tid] = ksum;
}

// ---------------- kernel 3: phi(q) -> out = q_p@kv / (q_p@k_sum) ----------------
__global__ __launch_bounds__(256) void perf_out_attn(
    const __bf16* __restrict__ qmat, const __bf16* __restrict__ rfb,
    const __bf16* __restrict__ kvws, const float* __restrict__ ksumws,
    __bf16* __restrict__ xattn) {
  int n0 = blockIdx.x * 32;
  int bh = blockIdx.y;
  int bb = bh / H_, hh = bh - bb * H_;
  int tid = threadIdx.x, w = tid >> 5, l = tid & 31;
  __shared__ __bf16 s_qp[32 * M_];  // [n_loc][m]
  __shared__ float  s_nsq[32];
  __shared__ float  s_norm[32];
  __shared__ float  s_part[256];
  const __bf16* qb = qmat + (size_t)bh * N_ * DH_;
  const __bf16* kvb = kvws + (size_t)bh * M_ * DH_;
  const float* ks = ksumws + (size_t)bh * M_;
  const float invSqrtM = 0.0625f;

  if (tid < 32) {
    float s = 0.0f;
    const __bf16* qr = qb + (size_t)(n0 + tid) * DH_;
    for (int d = 0; d < DH_; ++d) { float t = (float)qr[d]; s += t * t; }
    s_nsq[tid] = 0.5f * s;
  }
  __syncthreads();
#pragma unroll
  for (int ns = 0; ns < 2; ++ns) {
    bfrag16 a0 = load_a_bf16(qb + (size_t)(n0 + ns * 16) * DH_ + 0, DH_);
    bfrag16 a1 = load_a_bf16(qb + (size_t)(n0 + ns * 16) * DH_ + 32, DH_);
#pragma unroll
    for (int mi = 0; mi < 2; ++mi) {
      int mt = 2 * w + mi;
      bfrag16 b0 = load_bT(rfb + (size_t)(mt * 16) * DH_ + 0, DH_);
      bfrag16 b1 = load_bT(rfb + (size_t)(mt * 16) * DH_ + 32, DH_);
      v8f c = wmma_bf16(a0, b0, zero8());
      c = wmma_bf16(a1, b1, c);
      int rl = ns * 16 + (l >> 4) * 8;
      int mc = mt * 16 + (l & 15);
#pragma unroll
      for (int i = 0; i < 8; ++i) {
        float val = __expf(c[i] - s_nsq[rl + i]) * invSqrtM;
        s_qp[(size_t)(rl + i) * M_ + mc] = (__bf16)val;
      }
    }
  }
  __syncthreads();
  // normalizer: 8 threads per row, partial sums over 32 m each
  {
    int row = tid >> 3, p = tid & 7;
    const __bf16* rowp = s_qp + (size_t)row * M_ + p * 32;
    const float* ksp = ks + p * 32;
    float s = 0.0f;
#pragma unroll
    for (int m = 0; m < 32; ++m) s += (float)rowp[m] * ksp[m];
    s_part[tid] = s;
  }
  __syncthreads();
  if (tid < 32) {
    float s = 0.0f;
#pragma unroll
    for (int p = 0; p < 8; ++p) s += s_part[tid * 8 + p];
    s_norm[tid] = s;
  }
  __syncthreads();
  // one 16x16 out tile per wave: ns = w>>2, dt = w&3; K = 256
  v8f c = zero8();
  int ns = w >> 2, dt = w & 3;
  for (int mk = 0; mk < M_; mk += 32) {
    bfrag16 a = load_a_bf16(s_qp + (size_t)(ns * 16) * M_ + mk, M_);
    bfrag16 b = load_b_rm(kvb + (size_t)mk * DH_ + dt * 16, DH_);
    c = wmma_bf16(a, b, c);
  }
  int rl = ns * 16 + (l >> 4) * 8;
  int dcol = dt * 16 + (l & 15);
#pragma unroll
  for (int i = 0; i < 8; ++i) {
    int nloc = rl + i;
    float val = c[i] / (s_norm[nloc] + 1e-6f);
    xattn[((size_t)(bb * N_ + n0 + nloc)) * C_ + hh * DH_ + dcol] = (__bf16)val;
  }
}

// ---------------- kernel 4: proj + bias + residual + LayerNorm ----------------
__global__ __launch_bounds__(256) void perf_proj_ln(
    const __bf16* __restrict__ xattn, const __bf16* __restrict__ WtProj,
    const float* __restrict__ bproj, const float* __restrict__ xin,
    const float* __restrict__ gamma, const float* __restrict__ beta,
    float* __restrict__ out) {
  int r0 = blockIdx.x * 16;
  int tid = threadIdx.x, w = tid >> 5, l = tid & 31;
  __shared__ float s_y[16 * C_];   // 48 KB
  __shared__ float s_part[256];
  __shared__ float s_part2[256];
  __shared__ float s_mu[16], s_rstd[16];

  const __bf16* arow = xattn + (size_t)r0 * C_;
  const __bf16* Wc[6];
#pragma unroll
  for (int t = 0; t < 6; ++t) Wc[t] = WtProj + (size_t)((w * 6 + t) * 16) * C_;

  v8f acc[6];
#pragma unroll
  for (int t = 0; t < 6; ++t) acc[t] = zero8();

  // ping-pong double buffered K loop (C_ % 64 == 0)
  bfrag16 a0 = load_a_bf16(arow, C_);
  bfrag16 b0[6];
#pragma unroll
  for (int t = 0; t < 6; ++t) b0[t] = load_bT(Wc[t], C_);

  for (int kk = 0; kk < C_; kk += 64) {
    bfrag16 a1 = load_a_bf16(arow + kk + 32, C_);
    bfrag16 b1[6];
#pragma unroll
    for (int t = 0; t < 6; ++t) b1[t] = load_bT(Wc[t] + kk + 32, C_);
#pragma unroll
    for (int t = 0; t < 6; ++t) acc[t] = wmma_bf16(a0, b0[t], acc[t]);
    if (kk + 64 < C_) {
      a0 = load_a_bf16(arow + kk + 64, C_);
#pragma unroll
      for (int t = 0; t < 6; ++t) b0[t] = load_bT(Wc[t] + kk + 64, C_);
    }
#pragma unroll
    for (int t = 0; t < 6; ++t) acc[t] = wmma_bf16(a1, b1[t], acc[t]);
  }
  int rsub = (l >> 4) * 8, csub = l & 15;
#pragma unroll
  for (int t = 0; t < 6; ++t) {
    int col = (w * 6 + t) * 16 + csub;
#pragma unroll
    for (int i = 0; i < 8; ++i) {
      int row = rsub + i;
      s_y[row * C_ + col] =
          acc[t][i] + bproj[col] + xin[(size_t)(r0 + row) * C_ + col];
    }
  }
  __syncthreads();
  {
    int row = tid >> 4, p = tid & 15;
    float s = 0.0f, s2 = 0.0f;
    const float* yr = s_y + row * C_;
    for (int cidx = p * 48; cidx < p * 48 + 48; ++cidx) {
      float v = yr[cidx];
      s += v;
      s2 += v * v;
    }
    s_part[tid] = s;
    s_part2[tid] = s2;
  }
  __syncthreads();
  if (tid < 16) {
    float s = 0.0f, s2 = 0.0f;
#pragma unroll
    for (int p = 0; p < 16; ++p) { s += s_part[tid * 16 + p]; s2 += s_part2[tid * 16 + p]; }
    float mu = s * (1.0f / C_);
    float var = s2 * (1.0f / C_) - mu * mu;
    s_mu[tid] = mu;
    s_rstd[tid] = rsqrtf(var + 1e-5f);
  }
  __syncthreads();
  for (int idx = tid; idx < 16 * C_; idx += 256) {
    int row = idx / C_, col = idx - row * C_;
    out[(size_t)(r0 + row) * C_ + col] =
        (s_y[idx] - s_mu[row]) * s_rstd[row] * gamma[col] + beta[col];
  }
}

// ---------------- launch ----------------
extern "C" void kernel_launch(void* const* d_in, const int* in_sizes, int n_in,
                              void* d_out, int out_size, void* d_ws, size_t ws_size,
                              hipStream_t stream) {
  const float* x     = (const float*)d_in[0];
  const float* Wqkv  = (const float*)d_in[1];
  const float* Wproj = (const float*)d_in[2];
  const float* bproj = (const float*)d_in[3];
  const float* gamma = (const float*)d_in[4];
  const float* beta  = (const float*)d_in[5];
  const float* rf    = (const float*)d_in[6];
  float* out = (float*)d_out;
  (void)in_sizes; (void)n_in; (void)out_size; (void)ws_size;

  char* ws = (char*)d_ws;
  size_t off = 0;
  auto alloc = [&](size_t bytes) -> void* {
    off = (off + 255) & ~(size_t)255;
    void* p = ws + off;
    off += bytes;
    return p;
  };
  __bf16* WtQKV  = (__bf16*)alloc((size_t)3 * C_ * C_ * 2);
  __bf16* WtProj = (__bf16*)alloc((size_t)C_ * C_ * 2);
  __bf16* rfb    = (__bf16*)alloc((size_t)M_ * DH_ * 2);
  __bf16* qkv    = (__bf16*)alloc((size_t)3 * BH_ * N_ * DH_ * 2);
  __bf16* kvws   = (__bf16*)alloc((size_t)BH_ * M_ * DH_ * 2);
  float*  ksumws = (float*)alloc((size_t)BH_ * M_ * 4);
  __bf16* xattn  = (__bf16*)alloc((size_t)BN_ * C_ * 2);

  {
    int tot = 3 * C_ * C_;
    perf_transpose_bf16<<<(tot + 255) / 256, 256, 0, stream>>>(Wqkv, WtQKV, C_, 3 * C_);
  }
  {
    int tot = C_ * C_;
    perf_transpose_bf16<<<(tot + 255) / 256, 256, 0, stream>>>(Wproj, WtProj, C_, C_);
  }
  {
    int tot = M_ * DH_;
    perf_convert_bf16<<<(tot + 255) / 256, 256, 0, stream>>>(rf, rfb, tot);
  }

  perf_qkv_gemm<<<dim3((3 * C_) / 128, BN_ / 64), 256, 0, stream>>>(x, WtQKV, qkv);

  const __bf16* qm = qkv;
  const __bf16* km = qkv + (size_t)BH_ * N_ * DH_;
  const __bf16* vm = km + (size_t)BH_ * N_ * DH_;

  perf_kv_ksum<<<BH_, 256, 0, stream>>>(km, vm, rfb, kvws, ksumws);
  perf_out_attn<<<dim3(N_ / 32, BH_), 256, 0, stream>>>(qm, rfb, kvws, ksumws, xattn);
  perf_proj_ln<<<BN_ / 16, 256, 0, stream>>>(xattn, WtProj, bproj, x, gamma, beta, out);
}
